// SSD_47485158425057
// MI455X (gfx1250) — compile-verified, matrix operand order checked by reading, not run on hardware
//
#include <hip/hip_runtime.h>
#include <hip/hip_bf16.h>
#include <math.h>

// ---------------------------------------------------------------------------
// Mamba-2 SSD forward for MI455X (gfx1250), all matmuls via v_wmma_f32_16x16x32_bf16.
// Shapes: d=2048 h=32 p=n=64 chunk=256 c=16 b=2 t=4096 (BT = 8192 flattened rows).
// Workspace (f32 unless noted):
//   Bo[8192*64] Co[8192*64] dto[8192*32] acs[64*4096]
//   states[1024*4096] prevS[1024*4096] Ybf[8192*2048](bf16) Wbf[2048*2048](bf16)
//   total ~80 MB.
// ---------------------------------------------------------------------------

typedef __attribute__((ext_vector_type(16))) __bf16 v16bf;
typedef __attribute__((ext_vector_type(8)))  float  v8f;

#define WMMA_BF16(A, B, C) \
  __builtin_amdgcn_wmma_f32_16x16x32_bf16(false, (A), false, (B), (short)0, (C), false, false)

// K index for element e of a 16-bit 16x32 A/B fragment (ISA 7.12.2 layout):
// lanes 0-15: VGPR0-3 -> K 0..7, VGPR4-7 -> K 16..23; lanes 16-31: +8.
__device__ __forceinline__ int koff(int lane, int e) {
  return ((e < 8) ? e : e + 8) + ((lane & 16) ? 8 : 0);
}

// A fragment (16 rows x 32 K) from row-major f32, row stride ld (contiguous in k).
__device__ __forceinline__ v16bf load_a_f32(const float* src, int ld, int lane) {
  v16bf r; int m = lane & 15;
#pragma unroll
  for (int e = 0; e < 16; ++e) r[e] = (__bf16)src[m * ld + koff(lane, e)];
  return r;
}
// B fragment (32 K x 16 cols) where B_mat[k][col] = src[col*ld + k] (contiguous in k).
__device__ __forceinline__ v16bf load_bT_f32(const float* src, int ld, int lane) {
  v16bf r; int n = lane & 15;
#pragma unroll
  for (int e = 0; e < 16; ++e) r[e] = (__bf16)src[n * ld + koff(lane, e)];
  return r;
}
// bf16 LDS variants, both contiguous-in-k per lane -> vectorizable ds loads.
__device__ __forceinline__ v16bf load_a_bf16(const __bf16* src, int ld, int lane) {
  v16bf r; int m = lane & 15;
#pragma unroll
  for (int e = 0; e < 16; ++e) r[e] = src[m * ld + koff(lane, e)];
  return r;
}
__device__ __forceinline__ v16bf load_bT_bf16(const __bf16* src, int ld, int lane) {
  v16bf r; int n = lane & 15;
#pragma unroll
  for (int e = 0; e < 16; ++e) r[e] = src[n * ld + koff(lane, e)];
  return r;
}

__device__ __forceinline__ float softplus_fast(float v) {
  return (v > 20.f) ? v : __logf(1.f + __expf(v));
}

// ---------------------------------------------------------------------------
// Kernel 1: fused input projections.  out cols: [0,64)=B, [64,128)=C, [128,160)=dt.
// grid 640 blocks x 256 thr; wave job = 16x16 tile, K loop 2048 in steps of 32.
// (x + the 1.3 MB of weights are L2-resident on the 192 MB L2, so the per-wave
//  tiling here is L2-bound but cheap; this stage is only ~5.4 GFLOP.)
// ---------------------------------------------------------------------------
__global__ void ssd_proj(const float* __restrict__ x,
                         const float* __restrict__ B_w, const float* __restrict__ B_b,
                         const float* __restrict__ C_w, const float* __restrict__ C_b,
                         const float* __restrict__ dt_w, const float* __restrict__ dt_b,
                         float* __restrict__ Bo, float* __restrict__ Co,
                         float* __restrict__ dto) {
  const int lane = threadIdx.x & 31, wave = threadIdx.x >> 5;
  const int job = blockIdx.x * 8 + wave;        // 5120 jobs = 512 row-tiles * 10 col-tiles
  const int mt = job / 10, jt = job % 10;
  const int rm = mt * 16;
  const int col = lane & 15;
  const int jc = jt * 16 + col;
  const float* Wrow; float bias;
  if (jc < 64)       { Wrow = B_w  + (size_t)jc * 2048;         bias = B_b[jc]; }
  else if (jc < 128) { Wrow = C_w  + (size_t)(jc - 64) * 2048;  bias = C_b[jc - 64]; }
  else               { Wrow = dt_w + (size_t)(jc - 128) * 2048; bias = dt_b[jc - 128]; }

  v8f acc = {};
  for (int kb = 0; kb < 2048; kb += 32) {
    v16bf a = load_a_f32(x + (size_t)rm * 2048 + kb, 2048, lane);
    v16bf bf;
#pragma unroll
    for (int e = 0; e < 16; ++e) bf[e] = (__bf16)Wrow[kb + koff(lane, e)];
    acc = WMMA_BF16(a, bf, acc);
  }
  const int rofs = (lane & 16) ? 8 : 0;
#pragma unroll
  for (int v = 0; v < 8; ++v) {
    const int t = rm + rofs + v;
    float val = acc[v] + bias;
    if (jc < 64)        Bo[(size_t)t * 64 + jc] = val;
    else if (jc < 128)  Co[(size_t)t * 64 + (jc - 64)] = val;
    else                dto[(size_t)t * 32 + (jc - 128)] = softplus_fast(val);
  }
}

// ---------------------------------------------------------------------------
// Kernel 2: per-(b,h,chunk) cumulative sum of dA = dt * (-exp(A_log)).
// grid 1024 blocks x 256 thr.
// ---------------------------------------------------------------------------
__global__ void ssd_cumsum(const float* __restrict__ dto,
                           const float* __restrict__ A_log,
                           float* __restrict__ acs) {
  const int blk = blockIdx.x;
  const int ci = blk % 16, h = (blk / 16) % 32, b = blk / (16 * 32);
  const int tid = threadIdx.x;
  const int t = ci * 256 + tid;
  const float A = -__expf(A_log[h]);
  __shared__ float s[256];
  s[tid] = dto[((size_t)(b * 4096 + t)) * 32 + h] * A;
  __syncthreads();
#pragma unroll
  for (int off = 1; off < 256; off <<= 1) {
    float add = (tid >= off) ? s[tid - off] : 0.f;
    __syncthreads();
    s[tid] += add;
    __syncthreads();
  }
  acs[((size_t)(b * 32 + h)) * 4096 + t] = s[tid];
}

// ---------------------------------------------------------------------------
// Kernel 3: per-(b,chunk,h) block: Y_diag (+ D*x) and chunk states.
// grid 1024 blocks x 256 thr.  Writes Y in bf16 (final accumulate in yoff).
// LDS: xdt^T 32KB + stage 8KB + acs 1KB + decay 1KB.
// ---------------------------------------------------------------------------
__global__ void ssd_chunk(const float* __restrict__ x,
                          const float* __restrict__ Bo, const float* __restrict__ Co,
                          const float* __restrict__ dto, const float* __restrict__ acs,
                          const float* __restrict__ Dp,
                          __bf16* __restrict__ Ybf, float* __restrict__ states) {
  const int blk = blockIdx.x;
  const int h = blk % 32, ci = (blk / 32) % 16, b = blk / (32 * 16);
  const int lane = threadIdx.x & 31, wave = threadIdx.x >> 5;

  __shared__ __bf16 s_xdtT[64 * 256];      // xdt^T: [p][t], bf16
  __shared__ float  s_acs[256];            // per-chunk A cumsum
  __shared__ float  s_decay[256];          // exp(acs_last - acs[t])
  __shared__ __bf16 s_stage[8][16 * 32];   // per-wave bf16 score restage (D->A transpose)

  const int t0 = b * 4096 + ci * 256;      // flattened row base of this chunk

  { // cooperative load: one chunk-row per thread
    const int tr = threadIdx.x;
    const size_t abase = ((size_t)(b * 32 + h)) * 4096 + ci * 256;
    const float myacs = acs[abase + tr];
    const float alast = acs[abase + 255];
    const float dtv = dto[((size_t)(t0 + tr)) * 32 + h];
    const float* xr = x + ((size_t)(t0 + tr)) * 2048 + h * 64;
#pragma unroll 8
    for (int p = 0; p < 64; ++p) s_xdtT[p * 256 + tr] = (__bf16)(xr[p] * dtv);
    s_acs[tr] = myacs;
    s_decay[tr] = __expf(alast - myacs);
  }
  __syncthreads();

  const float Dh = Dp[h];
  const int colL = lane & 15, rofs = (lane & 16) ? 8 : 0;

  // ---- Y_diag: each wave owns 2 row-strips of 16 ----
  for (int si = 0; si < 2; ++si) {
    const int strip = wave * 2 + si;
    const int rbase = strip * 16;
    const float* Crow = Co + ((size_t)(t0 + rbase)) * 64;
    const v16bf ca0 = load_a_f32(Crow, 64, lane);       // K = n 0..31
    const v16bf ca1 = load_a_f32(Crow + 32, 64, lane);  // K = n 32..63
    v8f yacc[4];
#pragma unroll
    for (int pt = 0; pt < 4; ++pt) yacc[pt] = (v8f){};

    for (int jj = 0; jj < 8; ++jj) {
      if (jj * 2 > strip) break;            // fully above the causal diagonal
#pragma unroll
      for (int jh = 0; jh < 2; ++jh) {
        const int j = jj * 2 + jh;
        if (j <= strip) {
          const float* Brow = Bo + ((size_t)(t0 + j * 16)) * 64;
          v8f sc = {};
          sc = WMMA_BF16(ca0, load_bT_f32(Brow, 64, lane), sc);
          sc = WMMA_BF16(ca1, load_bT_f32(Brow + 32, 64, lane), sc);
          const int tcol = j * 16 + colL;
          const float aj = s_acs[tcol];
#pragma unroll
          for (int v = 0; v < 8; ++v) {
            const int trow = rbase + rofs + v;
            float lv = (trow >= tcol) ? sc[v] * __expf(s_acs[trow] - aj) : 0.f;
            s_stage[wave][(rofs + v) * 32 + jh * 16 + colL] = (__bf16)lv;
          }
        } else {
#pragma unroll
          for (int v = 0; v < 8; ++v)
            s_stage[wave][(rofs + v) * 32 + jh * 16 + colL] = (__bf16)0.f;
        }
      }
      // restaged scores as A operand (K = 32 chunk positions), xdt^T as B operand
      const v16bf sa = load_a_bf16(&s_stage[wave][0], 32, lane);
#pragma unroll
      for (int pt = 0; pt < 4; ++pt) {
        const v16bf xb = load_bT_bf16(s_xdtT + (size_t)(pt * 16) * 256 + jj * 32, 256, lane);
        yacc[pt] = WMMA_BF16(sa, xb, yacc[pt]);
      }
    }
    // epilogue: Y = Y_diag + D[h] * x  (bf16)
#pragma unroll
    for (int pt = 0; pt < 4; ++pt)
#pragma unroll
      for (int v = 0; v < 8; ++v) {
        const int trow = rbase + rofs + v;
        const size_t xi = ((size_t)(t0 + trow)) * 2048 + h * 64 + pt * 16 + colL;
        Ybf[xi] = (__bf16)(yacc[pt][v] + Dh * x[xi]);
      }
  }

  // ---- chunk states, oriented D[p][n] = xdt^T x (B * decay) ----
  for (int si = 0; si < 2; ++si) {
    const int tile = wave * 2 + si;
    const int pt = tile >> 2, nt = tile & 3;
    v8f acc = {};
    for (int kk = 0; kk < 8; ++kk) {
      const v16bf a = load_a_bf16(s_xdtT + (size_t)(pt * 16) * 256 + kk * 32, 256, lane);
      v16bf bf;
#pragma unroll
      for (int e = 0; e < 16; ++e) {
        const int tt = kk * 32 + koff(lane, e);
        bf[e] = (__bf16)(Bo[((size_t)(t0 + tt)) * 64 + nt * 16 + colL] * s_decay[tt]);
      }
      acc = WMMA_BF16(a, bf, acc);
    }
    const size_t sbase = ((size_t)((b * 16 + ci) * 32 + h)) * 4096;  // (b,c,h,p,n)
#pragma unroll
    for (int v = 0; v < 8; ++v)
      states[sbase + (size_t)(pt * 16 + rofs + v) * 64 + nt * 16 + colL] = acc[v];
  }
}

// ---------------------------------------------------------------------------
// Kernel 4: sequential inter-chunk state recurrence.  grid 64 blocks x 256 thr.
// prevS[c] = S_c;  S_{c+1} = exp(dAsum_c) * S_c + states_c.
// ---------------------------------------------------------------------------
__global__ void ssd_scan(const float* __restrict__ states,
                         const float* __restrict__ acs,
                         float* __restrict__ prevS) {
  const int bh = blockIdx.x;
  const int b = bh / 32, h = bh % 32;
  float run[16];
#pragma unroll
  for (int i = 0; i < 16; ++i) run[i] = 0.f;
  for (int ci = 0; ci < 16; ++ci) {
    const float f = __expf(acs[((size_t)(b * 32 + h)) * 4096 + ci * 256 + 255]);
    const size_t base = ((size_t)((b * 16 + ci) * 32 + h)) * 4096;
#pragma unroll
    for (int i = 0; i < 16; ++i) {
      const int e = threadIdx.x * 16 + i;
      prevS[base + e] = run[i];
      run[i] = run[i] * f + states[base + e];
    }
  }
}

// ---------------------------------------------------------------------------
// Kernel 5: Y += exp(acs[t]) * (C x prev_state), bf16 RMW.  grid 1024 x 256.
// ---------------------------------------------------------------------------
__global__ void ssd_yoff(const float* __restrict__ Co,
                         const float* __restrict__ prevS,
                         const float* __restrict__ acs,
                         __bf16* __restrict__ Ybf) {
  const int blk = blockIdx.x;
  const int h = blk % 32, ci = (blk / 32) % 16, b = blk / (32 * 16);
  const int lane = threadIdx.x & 31, wave = threadIdx.x >> 5;
  __shared__ __bf16 s_S[64 * 64];   // prev_state (p,n) in bf16
  __shared__ float  s_sdo[256];     // exp(acs[t])
  const size_t pbase = ((size_t)((b * 16 + ci) * 32 + h)) * 4096;
  for (int i = threadIdx.x; i < 4096; i += 256) s_S[i] = (__bf16)prevS[pbase + i];
  s_sdo[threadIdx.x] =
      __expf(acs[((size_t)(b * 32 + h)) * 4096 + ci * 256 + threadIdx.x]);
  __syncthreads();

  const int t0 = b * 4096 + ci * 256;
  const int colL = lane & 15, rofs = (lane & 16) ? 8 : 0;
  for (int si = 0; si < 2; ++si) {
    const int strip = wave * 2 + si, rbase = strip * 16;
    const float* Crow = Co + ((size_t)(t0 + rbase)) * 64;
    const v16bf ca0 = load_a_f32(Crow, 64, lane);
    const v16bf ca1 = load_a_f32(Crow + 32, 64, lane);
#pragma unroll
    for (int pt = 0; pt < 4; ++pt) {
      // Y[t][p] = sum_n C[t][n] * S[p][n]  -> contiguous-in-n read of (p,n) tile
      v8f acc = {};
      acc = WMMA_BF16(ca0, load_bT_bf16(s_S + (size_t)(pt * 16) * 64, 64, lane), acc);
      acc = WMMA_BF16(ca1, load_bT_bf16(s_S + (size_t)(pt * 16) * 64 + 32, 64, lane), acc);
#pragma unroll
      for (int v = 0; v < 8; ++v) {
        const int trow = rbase + rofs + v;
        const size_t yi = ((size_t)(t0 + trow)) * 2048 + h * 64 + pt * 16 + colL;
        Ybf[yi] = (__bf16)((float)Ybf[yi] + acc[v] * s_sdo[trow]);
      }
    }
  }
}

// ---------------------------------------------------------------------------
// Kernel 6a: one-shot out_w f32 -> bf16.  grid 16384 x 256.
// ---------------------------------------------------------------------------
__global__ void ssd_wcvt(const float* __restrict__ w, __bf16* __restrict__ wbf) {
  const size_t i = (size_t)blockIdx.x * 256 + threadIdx.x;
  wbf[i] = (__bf16)w[i];
}

// ---------------------------------------------------------------------------
// Kernel 6b: output projection (8192x2048)x(2048x2048), LDS-tiled.
// Block macro-tile 128x128 (grid 64x16 = 1024 blocks, 8 waves).
// Per K-step: stage 128x32 A and 128x32 B panels (pure 32B vector copies,
// operands already bf16), each wave = one 16-row strip x 8 col tiles
// -> 8x operand reuse on both sides, ~1 GB L2 traffic instead of 8 GB.
// ---------------------------------------------------------------------------
__global__ void ssd_outproj(const __bf16* __restrict__ Ybf,
                            const __bf16* __restrict__ Wbf,
                            const float* __restrict__ out_b,
                            float* __restrict__ out) {
  const int lane = threadIdx.x & 31, wave = threadIdx.x >> 5;
  const int mb = blockIdx.x >> 4;        // 64 row blocks
  const int nb = blockIdx.x & 15;        // 16 col blocks
  const int rm = mb * 128, cn = nb * 128;

  __shared__ __bf16 sA[128 * 32];        // [row][k]
  __shared__ __bf16 sB[128 * 32];        // [col][k]

  v8f acc[8];
#pragma unroll
  for (int ct = 0; ct < 8; ++ct) acc[ct] = (v8f){};

  const int tr = threadIdx.x >> 1;            // 0..127: row (A) / col (B)
  const int th = (threadIdx.x & 1) * 16;      // k-half: 0 or 16

  for (int kb = 0; kb < 2048; kb += 32) {
    __syncthreads();
    { // stage: each thread copies 16 bf16 (32 bytes) per panel, fully coalesced
      const int4* asrc = (const int4*)(Ybf + (size_t)(rm + tr) * 2048 + kb + th);
      const int4* bsrc = (const int4*)(Wbf + (size_t)(cn + tr) * 2048 + kb + th);
      int4* adst = (int4*)(sA + tr * 32 + th);
      int4* bdst = (int4*)(sB + tr * 32 + th);
      adst[0] = asrc[0]; adst[1] = asrc[1];
      bdst[0] = bsrc[0]; bdst[1] = bsrc[1];
    }
    __syncthreads();
    const v16bf a = load_a_bf16(sA + wave * 16 * 32, 32, lane);
#pragma unroll
    for (int ct = 0; ct < 8; ++ct) {
      const v16bf bfr = load_bT_bf16(sB + ct * 16 * 32, 32, lane);
      acc[ct] = WMMA_BF16(a, bfr, acc[ct]);
    }
  }

  const int colL = lane & 15, rofs = (lane & 16) ? 8 : 0;
#pragma unroll
  for (int ct = 0; ct < 8; ++ct) {
    const float bias = out_b[cn + ct * 16 + colL];
#pragma unroll
    for (int v = 0; v < 8; ++v)
      out[((size_t)(rm + wave * 16 + rofs + v)) * 2048 + cn + ct * 16 + colL] =
          acc[ct][v] + bias;
  }
}

// ---------------------------------------------------------------------------
extern "C" void kernel_launch(void* const* d_in, const int* in_sizes, int n_in,
                              void* d_out, int out_size, void* d_ws, size_t ws_size,
                              hipStream_t stream) {
  const float* x     = (const float*)d_in[0];
  const float* A_log = (const float*)d_in[1];
  const float* Dp    = (const float*)d_in[2];
  const float* B_w   = (const float*)d_in[3];
  const float* B_b   = (const float*)d_in[4];
  const float* C_w   = (const float*)d_in[5];
  const float* C_b   = (const float*)d_in[6];
  const float* dt_w  = (const float*)d_in[7];
  const float* dt_b  = (const float*)d_in[8];
  const float* out_w = (const float*)d_in[9];
  const float* out_b = (const float*)d_in[10];
  float* out = (float*)d_out;

  float*  Bo      = (float*)d_ws;                      //  8192*64
  float*  Co      = Bo      + (size_t)8192 * 64;       //  8192*64
  float*  dto     = Co      + (size_t)8192 * 64;       //  8192*32
  float*  acsw    = dto     + (size_t)8192 * 32;       //  64*4096
  float*  statesw = acsw    + (size_t)64 * 4096;       //  1024*4096
  float*  prevSw  = statesw + (size_t)1024 * 4096;     //  1024*4096
  __bf16* Ybf     = (__bf16*)(prevSw + (size_t)1024 * 4096);  // 8192*2048 bf16
  __bf16* Wbf     = Ybf + (size_t)8192 * 2048;                // 2048*2048 bf16

  ssd_proj   <<<  640, 256, 0, stream>>>(x, B_w, B_b, C_w, C_b, dt_w, dt_b, Bo, Co, dto);
  ssd_wcvt   <<<16384, 256, 0, stream>>>(out_w, Wbf);
  ssd_cumsum <<< 1024, 256, 0, stream>>>(dto, A_log, acsw);
  ssd_chunk  <<< 1024, 256, 0, stream>>>(x, Bo, Co, dto, acsw, Dp, Ybf, statesw);
  ssd_scan   <<<   64, 256, 0, stream>>>(statesw, acsw, prevSw);
  ssd_yoff   <<< 1024, 256, 0, stream>>>(Co, prevSw, acsw, Ybf);
  ssd_outproj<<< 1024, 256, 0, stream>>>(Ybf, Wbf, out_b, out);
}
